// MemoryUpdater_82927228551577
// MI455X (gfx1250) — matrix operand order repaired and checked by reading, not compiled, not run on hardware
//
#include <hip/hip_runtime.h>
#include <hip/hip_bf16.h>
#include <math.h>

// Problem constants (match reference: B=1024, N=1024, D=256)
#define BB   1024
#define NN   1024
#define DD   256
#define K3D  768          // 3*D
#define ROWS 2048         // 2*B message rows

typedef __attribute__((ext_vector_type(16))) __bf16 bf16x16;
typedef __attribute__((ext_vector_type(8)))  float  f32x8;

// ---------------------------------------------------------------------------
// WMMA tile helper: computes a 16x16 f32 tile of  A(MxK,row-major) * W^T
// where W is (N_out x K) row-major (i.e. out[m,n] = sum_k A[m,k]*W[n,k]).
// One wave per tile. Layouts per CDNA5 ISA 7.12.2 (16-bit A 16x32, B 32x16).
// ---------------------------------------------------------------------------
__device__ __forceinline__ f32x8 wmma_tile(const __bf16* __restrict__ A, int lda,
                                           const __bf16* __restrict__ W, int ldw,
                                           int m0, int n0, int K, f32x8 acc) {
    const int lane = threadIdx.x & 31;
    const int half = lane >> 4;   // 0: lanes 0-15, 1: lanes 16-31
    const int idx  = lane & 15;
    const __bf16* arow = A + (size_t)(m0 + idx) * lda;
    const __bf16* wrow = W + (size_t)(n0 + idx) * ldw;

    for (int k0 = 0; k0 < K; k0 += 32) {
        bf16x16 a, b;
        // A fragment: lanes<16 hold K {k0..k0+7, k0+16..k0+23}; lanes>=16 shift by 8
        const __bf16* a0 = arow + k0 + half * 8;        // elems 0..7  (VGPR0..3)
        const __bf16* a1 = a0 + 16;                     // elems 8..15 (VGPR4..7)
        #pragma unroll
        for (int i = 0; i < 8; ++i) { a[i] = a0[i]; a[i + 8] = a1[i]; }
        // B fragment: lane holds column n0+idx, K {k0..k0+7, k0+16..k0+23}; lanes>=16 shift by 8
        const __bf16* bp = wrow + k0 + half * 8;
        #pragma unroll
        for (int i = 0; i < 8; ++i) { b[i] = bp[i]; b[i + 8] = bp[i + 16]; }

        acc = __builtin_amdgcn_wmma_f32_16x16x32_bf16(
                  false, a, false, b, (short)0, acc, false, false);
    }
    return acc;
}

// ---------------------------------------------------------------------------
// Small utility kernels
// ---------------------------------------------------------------------------
__global__ void k_f32_to_bf16(const float* __restrict__ in, __bf16* __restrict__ out, int n) {
    int i = blockIdx.x * blockDim.x + threadIdx.x;
    if (i < n) out[i] = (__bf16)in[i];
}

__global__ void k_zero_f32(float* __restrict__ p, int n) {
    int i = blockIdx.x * blockDim.x + threadIdx.x;
    if (i < n) p[i] = 0.0f;
}

// Build gathered MLP input X (2048 x 768) in bf16.
// rows 0..1023  : [mem[src[b]], mem[tgt[b]], dt[b, src[b]]]
// rows 1024..   : [mem[tgt[b]], mem[src[b]], dt[b, tgt[b]]]
__global__ void k_gather_x(const float* __restrict__ mem, const float* __restrict__ dt,
                           const int* __restrict__ src, const int* __restrict__ tgt,
                           __bf16* __restrict__ X) {
    int t = blockIdx.x * blockDim.x + threadIdx.x;     // ROWS*K3D threads
    if (t >= ROWS * K3D) return;
    int r = t / K3D, c = t % K3D;
    int b = r & (BB - 1);
    bool is_src = r < BB;
    int i = is_src ? src[b] : tgt[b];
    int j = is_src ? tgt[b] : src[b];
    float v;
    if (c < DD)            v = mem[(size_t)i * DD + c];
    else if (c < 2 * DD)   v = mem[(size_t)j * DD + (c - DD)];
    else                   v = dt[(size_t)b * NN * DD + (size_t)i * DD + (c - 2 * DD)];
    X[t] = (__bf16)v;
}

// Segment counts (f32 so later division is direct)
__global__ void k_count(const int* __restrict__ src, const int* __restrict__ tgt,
                        float* __restrict__ cnt) {
    int r = blockIdx.x * blockDim.x + threadIdx.x;
    if (r < ROWS) {
        int node = (r < BB) ? src[r] : tgt[r - BB];
        atomicAdd(&cnt[node], 1.0f);
    }
}

// ---------------------------------------------------------------------------
// MLP layer 1:  H = relu(X @ W1^T + b1), per-half weights. Output bf16 2048x256.
// Tiles: 128 (M) x 16 (N) = 2048 tiles, 8 waves/block -> 256 blocks.
// ---------------------------------------------------------------------------
__global__ void k_msg_layer1(const __bf16* __restrict__ X,
                             const __bf16* __restrict__ w1s, const __bf16* __restrict__ w1t,
                             const float* __restrict__ b1s, const float* __restrict__ b1t,
                             __bf16* __restrict__ H) {
    const int wave = threadIdx.x >> 5;
    const int tile = blockIdx.x * 8 + wave;
    const int tn = tile & 15;          // DD/16 = 16 tiles along N
    const int tm = tile >> 4;
    const int m0 = tm * 16, n0 = tn * 16;
    const bool is_src = (m0 < BB);
    const __bf16* W    = is_src ? w1s : w1t;
    const float*  bias = is_src ? b1s : b1t;

    f32x8 acc = {};
    acc = wmma_tile(X, K3D, W, K3D, m0, n0, K3D, acc);

    const int lane = threadIdx.x & 31, half = lane >> 4, idx = lane & 15;
    const float bv = bias[n0 + idx];
    #pragma unroll
    for (int v = 0; v < 8; ++v) {
        float f = acc[v] + bv;
        f = f > 0.0f ? f : 0.0f;
        H[(size_t)(m0 + half * 8 + v) * DD + n0 + idx] = (__bf16)f;
    }
}

// ---------------------------------------------------------------------------
// MLP layer 2 + segment-sum scatter:  msg = H @ W2^T + b2 ; sums[node] += msg
// ---------------------------------------------------------------------------
__global__ void k_msg_layer2_scatter(const __bf16* __restrict__ H,
                                     const __bf16* __restrict__ w2s, const __bf16* __restrict__ w2t,
                                     const float* __restrict__ b2s, const float* __restrict__ b2t,
                                     const int* __restrict__ src, const int* __restrict__ tgt,
                                     float* __restrict__ sums) {
    const int wave = threadIdx.x >> 5;
    const int tile = blockIdx.x * 8 + wave;
    const int tn = tile & 15;
    const int tm = tile >> 4;
    const int m0 = tm * 16, n0 = tn * 16;
    const bool is_src = (m0 < BB);
    const __bf16* W    = is_src ? w2s : w2t;
    const float*  bias = is_src ? b2s : b2t;

    f32x8 acc = {};
    acc = wmma_tile(H, DD, W, DD, m0, n0, DD, acc);

    const int lane = threadIdx.x & 31, half = lane >> 4, idx = lane & 15;
    const float bv = bias[n0 + idx];
    #pragma unroll
    for (int v = 0; v < 8; ++v) {
        int r = m0 + half * 8 + v;
        int node = (r < BB) ? src[r] : tgt[r - BB];
        atomicAdd(&sums[(size_t)node * DD + n0 + idx], acc[v] + bv);
    }
}

// agg = sums / max(cnt,1); also produce bf16 copies of agg and memory
__global__ void k_agg(const float* __restrict__ sums, const float* __restrict__ cnt,
                      const float* __restrict__ mem,
                      __bf16* __restrict__ aggb, __bf16* __restrict__ memb) {
    int t = blockIdx.x * blockDim.x + threadIdx.x;     // NN*DD
    if (t >= NN * DD) return;
    int i = t >> 8;
    float a = sums[t] / fmaxf(cnt[i], 1.0f);
    aggb[t] = (__bf16)a;
    memb[t] = (__bf16)mem[t];
}

// ---------------------------------------------------------------------------
// GRU GEMM:  out(1024x768) = A(1024x256) @ W(768x256)^T + bias
// Tiles: 64 x 48 = 3072 tiles -> 384 blocks of 8 waves.
// ---------------------------------------------------------------------------
__global__ void k_gru_gemm(const __bf16* __restrict__ A, const __bf16* __restrict__ W,
                           const float* __restrict__ bias, float* __restrict__ out) {
    const int wave = threadIdx.x >> 5;
    const int tile = blockIdx.x * 8 + wave;
    const int tn = tile % 48;          // K3D/16 tiles along N
    const int tm = tile / 48;
    const int m0 = tm * 16, n0 = tn * 16;

    f32x8 acc = {};
    acc = wmma_tile(A, DD, W, DD, m0, n0, DD, acc);

    const int lane = threadIdx.x & 31, half = lane >> 4, idx = lane & 15;
    const float bv = bias[n0 + idx];
    #pragma unroll
    for (int v = 0; v < 8; ++v)
        out[(size_t)(m0 + half * 8 + v) * K3D + n0 + idx] = acc[v] + bv;
}

// GRU gates + touched-select -> final output
__global__ void k_gru_gate(const float* __restrict__ gi, const float* __restrict__ gh,
                           const float* __restrict__ mem, const float* __restrict__ cnt,
                           float* __restrict__ out) {
    int t = blockIdx.x * blockDim.x + threadIdx.x;     // NN*DD
    if (t >= NN * DD) return;
    int i = t >> 8, d = t & (DD - 1);
    size_t base = (size_t)i * K3D;
    float ir  = gi[base + d],           hr = gh[base + d];
    float iz  = gi[base + DD + d],      hz = gh[base + DD + d];
    float inn = gi[base + 2 * DD + d],  hn = gh[base + 2 * DD + d];
    float r = 1.0f / (1.0f + __expf(-(ir + hr)));
    float z = 1.0f / (1.0f + __expf(-(iz + hz)));
    float n = tanhf(inn + r * hn);
    float h = mem[t];
    float nh = (1.0f - z) * n + z * h;
    out[t] = (cnt[i] > 0.0f) ? nh : h;
}

// ---------------------------------------------------------------------------
// Launch
// ---------------------------------------------------------------------------
extern "C" void kernel_launch(void* const* d_in, const int* in_sizes, int n_in,
                              void* d_out, int out_size, void* d_ws, size_t ws_size,
                              hipStream_t stream) {
    const float* mem = (const float*)d_in[0];
    const int*   src = (const int*)  d_in[1];
    const int*   tgt = (const int*)  d_in[2];
    const float* dt  = (const float*)d_in[3];
    const float* sw1 = (const float*)d_in[4];
    const float* sb1 = (const float*)d_in[5];
    const float* sw2 = (const float*)d_in[6];
    const float* sb2 = (const float*)d_in[7];
    const float* tw1 = (const float*)d_in[8];
    const float* tb1 = (const float*)d_in[9];
    const float* tw2 = (const float*)d_in[10];
    const float* tb2 = (const float*)d_in[11];
    const float* wih = (const float*)d_in[12];
    const float* whh = (const float*)d_in[13];
    const float* bih = (const float*)d_in[14];
    const float* bhh = (const float*)d_in[15];
    float* out = (float*)d_out;

    // Workspace carve (256-byte aligned chunks)
    char* ws = (char*)d_ws;
    size_t off = 0;
    auto carve = [&](size_t bytes) { char* p = ws + off; off += (bytes + 255) & ~size_t(255); return p; };
    __bf16* Xb    = (__bf16*)carve((size_t)ROWS * K3D * 2);
    __bf16* Hb    = (__bf16*)carve((size_t)ROWS * DD  * 2);
    __bf16* w1sb  = (__bf16*)carve((size_t)DD  * K3D * 2);
    __bf16* w1tb  = (__bf16*)carve((size_t)DD  * K3D * 2);
    __bf16* w2sb  = (__bf16*)carve((size_t)DD  * DD  * 2);
    __bf16* w2tb  = (__bf16*)carve((size_t)DD  * DD  * 2);
    __bf16* wihb  = (__bf16*)carve((size_t)K3D * DD  * 2);
    __bf16* whhb  = (__bf16*)carve((size_t)K3D * DD  * 2);
    float*  sums  = (float*) carve((size_t)NN  * DD  * 4);
    float*  cnt   = (float*) carve((size_t)NN  * 4);
    __bf16* aggb  = (__bf16*)carve((size_t)NN  * DD  * 2);
    __bf16* memb  = (__bf16*)carve((size_t)NN  * DD  * 2);
    float*  gi    = (float*) carve((size_t)NN  * K3D * 4);
    float*  gh    = (float*) carve((size_t)NN  * K3D * 4);

    const int T = 256;
    auto blocks = [](int n, int t) { return (n + t - 1) / t; };

    // Weight conversions (f32 -> bf16)
    k_f32_to_bf16<<<blocks(DD * K3D, T), T, 0, stream>>>(sw1, w1sb, DD * K3D);
    k_f32_to_bf16<<<blocks(DD * K3D, T), T, 0, stream>>>(tw1, w1tb, DD * K3D);
    k_f32_to_bf16<<<blocks(DD * DD,  T), T, 0, stream>>>(sw2, w2sb, DD * DD);
    k_f32_to_bf16<<<blocks(DD * DD,  T), T, 0, stream>>>(tw2, w2tb, DD * DD);
    k_f32_to_bf16<<<blocks(K3D * DD, T), T, 0, stream>>>(wih, wihb, K3D * DD);
    k_f32_to_bf16<<<blocks(K3D * DD, T), T, 0, stream>>>(whh, whhb, K3D * DD);

    // Gather MLP inputs
    k_gather_x<<<blocks(ROWS * K3D, T), T, 0, stream>>>(mem, dt, src, tgt, Xb);

    // Zero sums + cnt (contiguous in ws? not guaranteed -> two launches)
    k_zero_f32<<<blocks(NN * DD, T), T, 0, stream>>>(sums, NN * DD);
    k_zero_f32<<<blocks(NN, T), T, 0, stream>>>(cnt, NN);
    k_count<<<blocks(ROWS, T), T, 0, stream>>>(src, tgt, cnt);

    // MLP layer 1 (WMMA): 2048 tiles / 8 waves = 256 blocks
    k_msg_layer1<<<256, T, 0, stream>>>(Xb, w1sb, w1tb, sb1, tb1, Hb);
    // MLP layer 2 + scatter (WMMA): 256 blocks
    k_msg_layer2_scatter<<<256, T, 0, stream>>>(Hb, w2sb, w2tb, sb2, tb2, src, tgt, sums);

    // Aggregate mean + bf16 conversions
    k_agg<<<blocks(NN * DD, T), T, 0, stream>>>(sums, cnt, mem, aggb, memb);

    // GRU GEMMs (WMMA): 3072 tiles / 8 = 384 blocks each
    k_gru_gemm<<<384, T, 0, stream>>>(aggb, wihb, bih, gi);
    k_gru_gemm<<<384, T, 0, stream>>>(memb, whhb, bhh, gh);

    // Gates + select
    k_gru_gate<<<blocks(NN * DD, T), T, 0, stream>>>(gi, gh, mem, cnt, out);
}
